// MACG_51127290692059
// MI455X (gfx1250) — matrix operand order repaired and checked by reading, not compiled, not run on hardware
//
#include <hip/hip_runtime.h>

// MACG mixture log-likelihood for MI455X (gfx1250), wave32.
// Big GEMM Z = S_k^T X in bf16 WMMA (f32 accum), Gram Y = Z^T Z in f32 WMMA,
// Cholesky logdet + logsumexp fused. Deterministic two-stage reduction.
//
// Round-2 changes:
//  - 4-slot / distance-3 ring buffer for A-operand tiles (hides L2 latency;
//    pipeline flows across k since tiles are contiguous in (k,rs,pt)).
//  - Gram relayout stores Z transposed: 2x ds_store_b128 + 1x ds_load_b64
//    per step instead of 16 scalar DS ops; stride 20 avoids bank conflicts.
//
// Workspace layout (needs ~2.1 MB):
//   [0, 2MB)   : pre-swizzled bf16 A-operand tiles of S_k^T
//   +2MB       : 33 floats of constants
//   +2MB+256B  : per-block partial sums (<= 1563 floats)

#define P_DIM 256
#define Q_DIM 8
#define K_MIX 16
#define TRI_N 32896 // 256*257/2
#define ZSTR  20    // padded LDS stride (floats): 16B-aligned, conflict-free

typedef __attribute__((ext_vector_type(16))) __bf16 v16bf;
typedef __attribute__((ext_vector_type(8)))  float  v8f;
typedef __attribute__((ext_vector_type(4)))  float  v4f;
typedef __attribute__((ext_vector_type(2)))  float  v2f;

static __device__ __forceinline__ unsigned short f2bf(float f) {
  unsigned int u = __float_as_uint(f);
  u += 0x7FFFu + ((u >> 16) & 1u);          // round-to-nearest-even
  return (unsigned short)(u >> 16);
}

static __device__ __forceinline__ v8f wmma_bf16(v16bf a, v16bf b, v8f c) {
  return __builtin_amdgcn_wmma_f32_16x16x32_bf16(false, a, false, b, (short)0, c, false, false);
}
static __device__ __forceinline__ v8f wmma_f32(v2f a, v2f b, v8f c) {
  return __builtin_amdgcn_wmma_f32_16x16x4_f32(false, a, false, b, (short)0, c, false, false);
}

struct AA { uint4 a, b; };

static __device__ __forceinline__ void lda(AA& d, const unsigned short* __restrict__ ST3,
                                           int g, int lane) {
  const uint4* p = (const uint4*)(ST3 + (((g << 5) + lane) << 4));
  d.a = p[0];
  d.b = p[1];
}
static __device__ __forceinline__ v16bf aav(const AA& x) {
  union { uint4 u[2]; v16bf v; } c;
  c.u[0] = x.a; c.u[1] = x.b;
  return c.v;
}

// ---------------------------------------------------------------------------
// Prep: scalar constants (log_det_S, log_softmax(pi), log surface area)
// ---------------------------------------------------------------------------
__global__ void macg_consts(const float* __restrict__ Svec,
                            const float* __restrict__ pi,
                            float* __restrict__ consts) {
  const double PI_D = 3.14159265358979323846;
  const int t = threadIdx.x;
  if (t < 16) {
    float s = 0.f;
    for (int r = 0; r < P_DIM; ++r)
      s += logf(fabsf(Svec[t * TRI_N + (r * (r + 1)) / 2 + r]));
    consts[t] = -2.f * s;                                   // log_det_S[k]
  } else if (t == 16) {
    float m = -3.4e38f;
    for (int k = 0; k < K_MIX; ++k) m = fmaxf(m, pi[k]);
    float se = 0.f;
    for (int k = 0; k < K_MIX; ++k) se += expf(pi[k] - m);
    const float lse = m + logf(se);
    for (int k = 0; k < K_MIX; ++k) consts[16 + k] = pi[k] - lse;  // logsoftmax
  } else if (t == 17) {
    double lg = 14.0 * log(PI_D);                           // q(q-1)/4 * ln(pi), q=8
    for (int i = 0; i < 8; ++i) lg += lgamma(128.0 - 0.5 * (double)i);
    consts[32] = (float)(lg - 8.0 * log(2.0) - 1024.0 * log(PI_D));  // LOGSA
  }
}

// ---------------------------------------------------------------------------
// Prep: scatter S_vec -> bf16 A-operand tiles in exact per-lane WMMA layout.
// Tile g = k*128 + pt*8 + rs is 16x32 of S_k^T: A[m][kk] = S_k[r][p],
// p = pt*16+m, r = rs*32+kk. Lane l holds 32B contiguous (two b128 loads).
// ---------------------------------------------------------------------------
__global__ void macg_prep(const float* __restrict__ Svec,
                          unsigned short* __restrict__ ST3) {
  const int idx  = blockIdx.x * 256 + threadIdx.x;  // < 1048576
  const int j    = idx & 7;
  const int half = (idx >> 3) & 1;
  const int l    = (idx >> 4) & 31;
  const int rs   = (idx >> 9) & 7;
  const int pt   = (idx >> 12) & 15;
  const int k    = idx >> 16;
  const int m    = l & 15;
  const int base = (l >> 4) * 8;
  const int kk   = half * 16 + base + j;
  const int r    = rs * 32 + kk;
  const int p    = pt * 16 + m;
  float v = 0.f;
  if (r >= p) v = Svec[k * TRI_N + (r * (r + 1)) / 2 + p];
  ST3[idx] = f2bf(v);
}

// ---------------------------------------------------------------------------
// Main: per block = 8 waves x 4 samples = 32 samples, all 16 components.
// ---------------------------------------------------------------------------
__global__ __launch_bounds__(256, 1) void macg_main(
    const float* __restrict__ X,
    const unsigned short* __restrict__ ST3,
    const float* __restrict__ consts,
    float* __restrict__ partials, int N) {
  __shared__ __attribute__((aligned(16))) float zbuf[8][3 * 16 * ZSTR];
  __shared__ float lp[32][K_MIX];     // log_pdf per sample-in-block per component
  __shared__ float red[32];

  const int tid  = threadIdx.x;
  const int w    = tid >> 5;
  const int lane = tid & 31;
  const int nlo  = lane & 15;
  const int hi   = lane >> 4;
  const long n0  = (long)blockIdx.x * 32;

  // Stage this wave's X slice in VGPRs as bf16 B-operands (reused for all k).
  // B layout (32x16, 16-bit): lane col = nlo; elems e<8 -> K = hi*8 + e, e>=8 -> K = 16 + hi*8 + (e-8).
  v16bf Breg[2][8];
  #pragma unroll
  for (int ct = 0; ct < 2; ++ct) {
    const int c = ct * 16 + nlo;               // column within wave's 32
    long n = n0 + w * 4 + (c >> 3);
    if (n >= N) n = N - 1;
    const float* xp = X + n * 2048 + (c & 7);
    #pragma unroll
    for (int rs = 0; rs < 8; ++rs) {
      union { unsigned int u[8]; v16bf v; } bb;
      #pragma unroll
      for (int j = 0; j < 8; ++j) {
        const int kk = (j >> 2) * 16 + hi * 8 + 2 * (j & 3);
        const float f0 = xp[(rs * 32 + kk) * 8];
        const float f1 = xp[(rs * 32 + kk + 1) * 8];
        bb.u[j] = (unsigned int)f2bf(f0) | ((unsigned int)f2bf(f1) << 16);
      }
      Breg[ct][rs] = bb.v;
    }
  }

  const v8f vzero = {0.f, 0.f, 0.f, 0.f, 0.f, 0.f, 0.f, 0.f};

  // A-operand ring buffer: 4 slots, prefetch distance 3.
  AA ring[4];
  #pragma unroll
  for (int i = 0; i < 3; ++i) lda(ring[i], ST3, i, lane);

  #pragma unroll 1
  for (int k = 0; k < K_MIX; ++k) {
    v8f acc[16][2];
    #pragma unroll
    for (int pt = 0; pt < 16; ++pt) { acc[pt][0] = vzero; acc[pt][1] = vzero; }

    // Z = S_k^T X : stream A tiles through the ring (loads 3 tiles ahead).
    #pragma unroll
    for (int rs = 0; rs < 8; ++rs) {
      #pragma unroll
      for (int pt = 0; pt < 16; ++pt) {
        int g = k * 128 + rs * 16 + pt + 3;     // next-tile prefetch (uniform)
        if (g > 2047) g = 2047;
        lda(ring[(pt + 3) & 3], ST3, g, lane);
        const v16bf a = aav(ring[pt & 3]);
        acc[pt][0] = wmma_bf16(a, Breg[0][rs], acc[pt][0]);
        acc[pt][1] = wmma_bf16(a, Breg[1][rs], acc[pt][1]);
      }
    }

    // Y = Z^T Z per 16-col tile: store Z transposed (2x ds_store_b128),
    // reload as f32 WMMA operands (1x ds_load_b64 per step); A==B operands.
    #pragma unroll
    for (int ct = 0; ct < 2; ++ct) {
      v8f y = vzero;
      #pragma unroll
      for (int pt = 0; pt < 16; ++pt) {
        float* zp = &zbuf[w][nlo * ZSTR + hi * 8];
        *(v4f*)zp       = __builtin_shufflevector(acc[pt][ct], acc[pt][ct], 0, 1, 2, 3);
        *(v4f*)(zp + 4) = __builtin_shufflevector(acc[pt][ct], acc[pt][ct], 4, 5, 6, 7);
        #pragma unroll
        for (int ps = 0; ps < 4; ++ps) {
          const v2f op = *(const v2f*)&zbuf[w][nlo * ZSTR + ps * 4 + 2 * hi];
          y = wmma_f32(op, op, y);
        }
      }
      float* yp = &zbuf[w][(16 + ct * 16 + nlo) * ZSTR + hi * 8];
      *(v4f*)yp       = __builtin_shufflevector(y, y, 0, 1, 2, 3);
      *(v4f*)(yp + 4) = __builtin_shufflevector(y, y, 4, 5, 6, 7);
    }

    // 8x8 Cholesky log|det| : lanes 0..3 handle the wave's 4 samples.
    // (Y stored transposed, but Y is symmetric.)
    if (lane < 4) {
      const int ct = lane >> 1, ss = lane & 1;
      const float* yb = &zbuf[w][(16 + ct * 16) * ZSTR];
      float A8[8][8];
      #pragma unroll
      for (int i = 0; i < 8; ++i)
        #pragma unroll
        for (int j = 0; j <= i; ++j)
          A8[i][j] = yb[(8 * ss + i) * ZSTR + 8 * ss + j];
      float logdet = 0.f;
      #pragma unroll
      for (int j = 0; j < 8; ++j) {
        float d = A8[j][j];
        #pragma unroll
        for (int t = 0; t < j; ++t) d -= A8[j][t] * A8[j][t];
        d = fmaxf(d, 1e-30f);
        logdet += logf(d);                    // sum log d_j == log det
        const float rinv = 1.0f / sqrtf(d);
        #pragma unroll
        for (int i = j + 1; i < 8; ++i) {
          float s = A8[i][j];
          #pragma unroll
          for (int t = 0; t < j; ++t) s -= A8[i][t] * A8[j][t];
          A8[i][j] = s * rinv;
        }
      }
      lp[w * 4 + lane][k] = logdet;
    }
  }

  __syncthreads();

  // density = LOGSA - 4*log_det_S[k] - 128*log_pdf + logsoftmax(pi)[k];
  // per-sample logsumexp over k, then block partial sum (fixed order).
  if (tid < 32) {
    const long n = n0 + tid;
    float val = 0.f;
    if (n < N) {
      const float logsa = consts[32];
      float dens[K_MIX];
      float m = -3.4e38f;
      #pragma unroll
      for (int k = 0; k < K_MIX; ++k) {
        dens[k] = logsa - 4.0f * consts[k] - 128.0f * lp[tid][k] + consts[16 + k];
        m = fmaxf(m, dens[k]);
      }
      float se = 0.f;
      #pragma unroll
      for (int k = 0; k < K_MIX; ++k) se += expf(dens[k] - m);
      val = m + logf(se);
    }
    red[tid] = val;
  }
  __syncthreads();
  if (tid == 0) {
    float s = 0.f;
    for (int i = 0; i < 32; ++i) s += red[i];
    partials[blockIdx.x] = s;
  }
}

// ---------------------------------------------------------------------------
// Deterministic final reduction (single block, fixed tree).
// ---------------------------------------------------------------------------
__global__ void macg_reduce(const float* __restrict__ partials, int nb,
                            float* __restrict__ out) {
  __shared__ float sh[256];
  float s = 0.f;
  for (int i = threadIdx.x; i < nb; i += 256) s += partials[i];
  sh[threadIdx.x] = s;
  __syncthreads();
  for (int st = 128; st > 0; st >>= 1) {
    if ((int)threadIdx.x < st) sh[threadIdx.x] += sh[threadIdx.x + st];
    __syncthreads();
  }
  if (threadIdx.x == 0) out[0] = sh[0];
}

extern "C" void kernel_launch(void* const* d_in, const int* in_sizes, int n_in,
                              void* d_out, int out_size, void* d_ws, size_t ws_size,
                              hipStream_t stream) {
  const float* X    = (const float*)d_in[0];
  const float* Svec = (const float*)d_in[1];
  const float* pi   = (const float*)d_in[2];
  float* out = (float*)d_out;

  const int N = in_sizes[0] / (P_DIM * Q_DIM);

  unsigned short* st3 = (unsigned short*)d_ws;                       // 2 MB
  float* consts  = (float*)((char*)d_ws + (size_t)2 * 1024 * 1024);  // 33 floats
  float* partial = consts + 64;

  const int nb = (N + 31) / 32;

  macg_consts<<<1, 64, 0, stream>>>(Svec, pi, consts);
  macg_prep<<<4096, 256, 0, stream>>>(Svec, st3);
  macg_main<<<nb, 256, 0, stream>>>(X, st3, consts, partial, N);
  macg_reduce<<<1, 256, 0, stream>>>(partial, nb, out);
}
